// FluidLayer2D_6055903888136
// MI455X (gfx1250) — matrix operand order repaired and checked by reading, not compiled, not run on hardware
//
#include <hip/hip_runtime.h>
#include <math.h>

// ---------------- problem constants ----------------
#define B_     4
#define C_     64
#define HID_   128
#define H_     192
#define W_     192
#define HW_    (H_ * W_)            // 36864
#define NROWS_ (B_ * HW_)           // 147456
#define NTOT_  (B_ * C_ * HW_)      // 9437184
#define NPROBE_ (B_ * C_ * 8 * 8)   // 16384

typedef __attribute__((ext_vector_type(16))) _Float16 v16h;
typedef __attribute__((ext_vector_type(8)))  _Float16 v8h;
typedef __attribute__((ext_vector_type(8)))  float    v8f;

// ---------------- small utility kernels ----------------
__global__ void zero_kernel(float* __restrict__ p, int n) {
  int i = blockIdx.x * 256 + threadIdx.x;
  if (i < n) p[i] = 0.0f;
}

// Pack W1 [64x128] and W2 [128x64] (f32, K-major) into f16, transposed so each
// B fragment (16 contiguous K values for one output column) is one 32B load.
//  w1t[col][k] : col 0..127, k 0..63   (8192 halves)
//  w2t[col][k] : col 0..63,  k 0..127  (8192 halves)
__global__ void pack_weights_kernel(const float* __restrict__ w1,
                                    const float* __restrict__ w2,
                                    _Float16* __restrict__ w1t,
                                    _Float16* __restrict__ w2t) {
  const int i = blockIdx.x * 256 + threadIdx.x;   // 0..8191
  if (i < HID_ * C_) {
    { const int col = i >> 6, k = i & 63;  w1t[i] = (_Float16)w1[k * HID_ + col]; }
    { const int col = i >> 7, k = i & 127; w2t[i] = (_Float16)w2[k * C_ + col]; }
  }
}

// f16 mirror of the ORIGINAL input u in [pixel][channel] (A-operand) order.
__launch_bounds__(256)
__global__ void u16_init_kernel(const float* __restrict__ u, _Float16* __restrict__ u16) {
  const int gid = blockIdx.x * 256 + threadIdx.x;   // pixel
  const int bb = gid / HW_;
  const int p  = gid - bb * HW_;
  const size_t base = (size_t)bb * (C_ * HW_) + p;
  for (int c = 0; c < C_; ++c)
    u16[(size_t)gid * C_ + c] = (_Float16)u[base + (size_t)c * HW_];
}

// one block per (b,c) plane: 4x4 avg pool + global mean
__launch_bounds__(256)
__global__ void pool_kernel(const float* __restrict__ uin,
                            float* __restrict__ pooled,
                            float* __restrict__ local4) {
  __shared__ float red[256];
  __shared__ float cellsum[16];
  const int bc = blockIdx.x;
  const float* up = uin + (size_t)bc * HW_;
  const int cellI = threadIdx.x >> 4;   // 0..15
  const int sub   = threadIdx.x & 15;   // 0..15
  const int ci = cellI >> 2, cj = cellI & 3;
  float s = 0.0f;
  const int y0 = ci * 48 + sub * 3;     // 3 rows x 48 px per thread
  for (int ry = 0; ry < 3; ++ry) {
    const float* row = up + (y0 + ry) * W_ + cj * 48;
    for (int xx = 0; xx < 48; ++xx) s += row[xx];
  }
  red[threadIdx.x] = s;
  __syncthreads();
  if (sub == 0) {
    float t = 0.0f;
    for (int i = 0; i < 16; ++i) t += red[cellI * 16 + i];
    cellsum[cellI] = t;
    local4[bc * 16 + cellI] = t * (1.0f / 2304.0f);
  }
  __syncthreads();
  if (threadIdx.x == 0) {
    float t = 0.0f;
    for (int i = 0; i < 16; ++i) t += cellsum[i];
    pooled[bc] = t * (1.0f / (float)HW_);
  }
}

// one block per (b,c) plane: 8x8 avg pool of the ORIGINAL u -> prev_probe
__launch_bounds__(256)
__global__ void probe_init_kernel(const float* __restrict__ uin,
                                  float* __restrict__ prevp) {
  __shared__ float red[256];
  const int bc = blockIdx.x;
  const float* up = uin + (size_t)bc * HW_;
  const int cellI = threadIdx.x >> 2;   // 0..63
  const int sub   = threadIdx.x & 3;    // 0..3
  const int ci = cellI >> 3, cj = cellI & 7;
  float s = 0.0f;
  const int y0 = ci * 24 + sub * 6;     // 6 rows x 24 px per thread
  for (int ry = 0; ry < 6; ++ry) {
    const float* row = up + (y0 + ry) * W_ + cj * 24;
    for (int xx = 0; xx < 24; ++xx) s += row[xx];
  }
  red[threadIdx.x] = s;
  __syncthreads();
  if (sub == 0) {
    float t = red[cellI * 4] + red[cellI * 4 + 1] + red[cellI * 4 + 2] + red[cellI * 4 + 3];
    prevp[bc * 64 + cellI] = t * (1.0f / 576.0f);
  }
}

// single block: h_state pump + 1x1 conv on pooled 4x4 features
__launch_bounds__(256)
__global__ void state_kernel(const float* __restrict__ pooled,
                             const float* __restrict__ gw, const float* __restrict__ gb,
                             const float* __restrict__ vw, const float* __restrict__ vb,
                             const float* __restrict__ lw, const float* __restrict__ lb,
                             const float* __restrict__ local4,
                             float* __restrict__ hstate, float* __restrict__ lconv) {
  const int t = threadIdx.x;            // 256 = 4 batches x 64 channels
  const int b = t >> 6, c = t & 63;
  float g = gb[c], v = vb[c];
  for (int i = 0; i < 64; ++i) {
    float pv = pooled[b * 64 + i];
    g += pv * gw[i * 64 + c];           // pooled @ gate_w
    v += pv * vw[i * 64 + c];           // pooled @ value_w
  }
  g = 1.0f / (1.0f + expf(-g));
  v = tanhf(v);
  hstate[t] += g * v;

  // local_conv[b][c][cell] = local_b[c] + sum_in local_w[c][in]*local4[b][in][cell]
  float accv[16];
  for (int k = 0; k < 16; ++k) accv[k] = lb[c];
  for (int i = 0; i < 64; ++i) {
    float wv = lw[c * 64 + i];
    const float* l4 = local4 + (b * 64 + i) * 16;
    for (int k = 0; k < 16; ++k) accv[k] += wv * l4[k];
  }
  for (int k = 0; k < 16; ++k) lconv[(b * 64 + c) * 16 + k] = accv[k];
}

// ---------------- WMMA MLP: react = gelu(X @ W1 + b1) @ W2 + b2 ----------------
// Each wave handles 16 pixel rows (M=16).  Block = 8 waves = 128 rows.
// A-fragment layout (16-bit 16x32, contiguous K, same as B): lane m = lane&15,
//   kb8 = (lane>>4)*16; halves 0..7 -> K=kb8+i ; halves 8..15 -> K=kb8+8+i.
//   u16 is [row][K] so each half-fragment is one contiguous 16B load.
// B-fragment layout (16-bit 32x16, contiguous K order used here, matching A):
//   lane n = lane&15; K = (lane>>4)*16 + half_index -> one contiguous 32B load
//   from the pre-transposed f16 weights.
// C/D layout: lane n = lane&15; VGPR v -> row M = (lane>>4)*8 + v.
__launch_bounds__(256)
__global__ void react_kernel(const _Float16* __restrict__ u16,
                             const _Float16* __restrict__ w1t, const float* __restrict__ b1,
                             const _Float16* __restrict__ w2t, const float* __restrict__ b2,
                             float* __restrict__ react) {
  __shared__ _Float16 ldsH[8][16][HID_];         // 32 KB: per-wave gelu(H1) tile
  const int lane = threadIdx.x & 31;
  const int wave = threadIdx.x >> 5;
  const int rowBase = blockIdx.x * 128 + wave * 16;
  const int m   = lane & 15;
  const int kb8 = (lane >> 4) * 16;              // A half-wave K base
  const int kof = (lane >> 4) * 16;              // B half-wave K base
  const int n   = lane & 15;

  // ---- A fragments: 4 x 16B vector loads ----
  const _Float16* uR = u16 + (size_t)(rowBase + m) * C_;
  const v8h t0 = *(const v8h*)(uR + kb8);        // K =  kb8 .. kb8+7
  const v8h t1 = *(const v8h*)(uR + 8 + kb8);    // K = 8+kb8 ..
  const v8h t2 = *(const v8h*)(uR + 32 + kb8);   // K = 32+kb8 ..
  const v8h t3 = *(const v8h*)(uR + 40 + kb8);   // K = 40+kb8 ..
  v16h a0, a1;
#pragma unroll
  for (int i = 0; i < 8; ++i) {
    a0[i] = t0[i]; a0[8 + i] = t1[i];
    a1[i] = t2[i]; a1[8 + i] = t3[i];
  }

  // ---- GEMM1: [16x64] @ [64x128], 8 column tiles ----
#pragma unroll
  for (int ct = 0; ct < 8; ++ct) {
    const int col = ct * 16 + n;
    const v16h bf0 = *(const v16h*)(w1t + col * C_ + kof);        // K = 0..31 slice
    const v16h bf1 = *(const v16h*)(w1t + col * C_ + 32 + kof);   // K = 32..63 slice
    v8f acc;
    const float bias = b1[col];
#pragma unroll
    for (int v = 0; v < 8; ++v) acc[v] = bias;
    acc = __builtin_amdgcn_wmma_f32_16x16x32_f16(false, a0, false, bf0, (short)0, acc, false, false);
    acc = __builtin_amdgcn_wmma_f32_16x16x32_f16(false, a1, false, bf1, (short)0, acc, false, false);
    // exact GELU, stash as f16 in LDS (cross-lane transpose to A layout)
#pragma unroll
    for (int v = 0; v < 8; ++v) {
      float xg = acc[v];
      float gl = 0.5f * xg * (1.0f + erff(xg * 0.7071067811865475f));
      const int mrow = (lane >> 4) * 8 + v;
      ldsH[wave][mrow][col] = (_Float16)gl;
    }
  }
  __syncthreads();

  // ---- GEMM2: [16x128] @ [128x64], 4 column tiles ----
  v16h a2[4];
#pragma unroll
  for (int kb = 0; kb < 4; ++kb) {
    const v8h l0 = *(const v8h*)&ldsH[wave][m][kb * 32 + kb8];
    const v8h l1 = *(const v8h*)&ldsH[wave][m][kb * 32 + 8 + kb8];
#pragma unroll
    for (int i = 0; i < 8; ++i) { a2[kb][i] = l0[i]; a2[kb][8 + i] = l1[i]; }
  }
#pragma unroll
  for (int ct = 0; ct < 4; ++ct) {
    const int col = ct * 16 + n;
    v8f acc;
    const float bias = b2[col];
#pragma unroll
    for (int v = 0; v < 8; ++v) acc[v] = bias;
#pragma unroll
    for (int kb = 0; kb < 4; ++kb) {
      const v16h bf = *(const v16h*)(w2t + col * HID_ + kb * 32 + kof);
      acc = __builtin_amdgcn_wmma_f32_16x16x32_f16(false, a2[kb], false, bf, (short)0, acc, false, false);
    }
#pragma unroll
    for (int v = 0; v < 8; ++v) {
      const int mrow = (lane >> 4) * 8 + v;
      react[(size_t)(rowBase + mrow) * C_ + col] = acc[v];
    }
  }
}

// ---------------- fused combine: laplacian + du + u_cand + norm + stats ----------------
__launch_bounds__(256)
__global__ void combine_kernel(const float* __restrict__ uin,
                               const float* __restrict__ react,
                               const float* __restrict__ hstate,
                               const float* __restrict__ lconv,
                               const float* __restrict__ norm_w,
                               const float* __restrict__ alpha_g,
                               const float* __restrict__ alpha_l,
                               const float* __restrict__ log_dt,
                               const float* __restrict__ diff_w,
                               float* __restrict__ uout,
                               _Float16* __restrict__ u16out,
                               float* __restrict__ curp,
                               float* __restrict__ acc,
                               int step, int do_norm) {
  const int gid = blockIdx.x * 256 + threadIdx.x;   // one pixel per thread
  const int bb = gid / HW_;
  const int p  = gid - bb * HW_;
  const int y = p / W_;
  const int x = p - y * W_;

  float dt = expf(log_dt[0]);
  dt = fminf(fmaxf(dt, 0.005f), 0.35f);
  const float ag = log1pf(expf(alpha_g[0]));        // softplus
  const float al = log1pf(expf(alpha_l[0]));
  const float dw0 = 0.25f * diff_w[0];
  const float dw1 = 0.25f * diff_w[1];
  const float dw2 = 0.25f * diff_w[2];

  // bilinear 4 -> 192 (align_corners=False, clipped)
  float sy = (y + 0.5f) * (1.0f / 48.0f) - 0.5f; sy = fminf(fmaxf(sy, 0.0f), 3.0f);
  float sx = (x + 0.5f) * (1.0f / 48.0f) - 0.5f; sx = fminf(fmaxf(sx, 0.0f), 3.0f);
  const int i0 = (int)sy; const int i1 = (i0 + 1 < 3) ? i0 + 1 : 3; const float ty = sy - (float)i0;
  const int j0 = (int)sx; const int j1 = (j0 + 1 < 3) ? j0 + 1 : 3; const float tx = sx - (float)j0;
  const float w00 = (1.0f - ty) * (1.0f - tx), w01 = (1.0f - ty) * tx;
  const float w10 = ty * (1.0f - tx),          w11 = ty * tx;

  const int cell = (y / 24) * 8 + (x / 24);          // 8x8 probe cell
  const size_t base = (size_t)bb * (C_ * HW_) + p;
  _Float16* u16R = u16out + (size_t)gid * C_;

  float duabs = 0.0f, lapabs = 0.0f, ss = 0.0f;
  for (int c = 0; c < C_; ++c) {
    const float* up = uin + base + (size_t)c * HW_;  // up[0] = center
    const float ctr = up[0];
    // zero-padded 5-point stencil at dilations 1,4,16
    float s1  = (y >= 1  ? up[-W_]      : 0.0f) + (y <= 190 ? up[W_]      : 0.0f)
              + (x >= 1  ? up[-1]       : 0.0f) + (x <= 190 ? up[1]       : 0.0f);
    float s4  = (y >= 4  ? up[-4 * W_]  : 0.0f) + (y <= 187 ? up[4 * W_]  : 0.0f)
              + (x >= 4  ? up[-4]       : 0.0f) + (x <= 187 ? up[4]       : 0.0f);
    float s16 = (y >= 16 ? up[-16 * W_] : 0.0f) + (y <= 175 ? up[16 * W_] : 0.0f)
              + (x >= 16 ? up[-16]      : 0.0f) + (x <= 175 ? up[16]      : 0.0f);
    const float lap = dw0 * (s1 - 4.0f * ctr) + dw1 * (s4 - 4.0f * ctr) + dw2 * (s16 - 4.0f * ctr);

    const float* lc = lconv + ((bb * 64 + c) << 4);
    const float lv = w00 * lc[i0 * 4 + j0] + w01 * lc[i0 * 4 + j1]
                   + w10 * lc[i1 * 4 + j0] + w11 * lc[i1 * 4 + j1];

    const float du = lap + react[(size_t)gid * C_ + c] + ag * hstate[bb * 64 + c] + al * lv;
    const float cd = ctr + dt * du;
    duabs  += fabsf(du);
    lapabs += fabsf(lap);
    ss     += cd * cd;
    uout[base + (size_t)c * HW_] = cd;               // u_cand (pre-norm)
    if (!do_norm) u16R[c] = (_Float16)cd;            // f16 mirror for next react
    atomicAdd(&curp[(bb * 64 + c) * 64 + cell], cd); // probe sums over u_cand
  }
  if (do_norm) {
    const float inv = 1.0f / sqrtf(ss * (1.0f / 64.0f) + 1e-8f);
    for (int c = 0; c < C_; ++c) {
      const size_t idx = base + (size_t)c * HW_;
      const float un = uout[idx] * inv * norm_w[c];  // second pass: rmsnorm
      uout[idx] = un;
      u16R[c] = (_Float16)un;
    }
  }
  // block-reduce |du| and |lap| sums -> one atomic per block
  __shared__ float rn[256], rd[256];
  rn[threadIdx.x] = duabs; rd[threadIdx.x] = lapabs;
  __syncthreads();
  for (int o = 128; o > 0; o >>= 1) {
    if (threadIdx.x < o) { rn[threadIdx.x] += rn[threadIdx.x + o]; rd[threadIdx.x] += rd[threadIdx.x + o]; }
    __syncthreads();
  }
  if (threadIdx.x == 0) { atomicAdd(&acc[step], rn[0]); atomicAdd(&acc[6 + step], rd[0]); }
}

// probe turbulence vs previous probe; rotate state
__launch_bounds__(256)
__global__ void probe_finish_kernel(const float* __restrict__ curp,
                                    float* __restrict__ prevp,
                                    float* __restrict__ acc, int step) {
  const int i = blockIdx.x * 256 + threadIdx.x;     // 16384 cells
  float num = 0.0f, den = 0.0f;
  if (i < NPROBE_) {
    const float cur  = curp[i] * (1.0f / 576.0f);
    const float prev = prevp[i];
    num = fabsf(cur - prev);
    den = fabsf(prev);
    prevp[i] = cur;
  }
  __shared__ float rn[256], rd[256];
  rn[threadIdx.x] = num; rd[threadIdx.x] = den;
  __syncthreads();
  for (int o = 128; o > 0; o >>= 1) {
    if (threadIdx.x < o) { rn[threadIdx.x] += rn[threadIdx.x + o]; rd[threadIdx.x] += rd[threadIdx.x + o]; }
    __syncthreads();
  }
  if (threadIdx.x == 0) { atomicAdd(&acc[12 + step], rn[0]); atomicAdd(&acc[18 + step], rd[0]); }
}

__global__ void finalize_kernel(const float* __restrict__ acc, float* __restrict__ out2) {
  if (threadIdx.x == 0 && blockIdx.x == 0) {
    float dturb = 0.0f, senergy = 0.0f;
    for (int s = 0; s < 6; ++s) {
      const float se   = acc[s]      * (1.0f / (float)NTOT_);
      const float le   = acc[6 + s]  * (1.0f / (float)NTOT_);
      const float num  = acc[12 + s] * (1.0f / (float)NPROBE_);
      const float den  = acc[18 + s] * (1.0f / (float)NPROBE_);
      const float stop = num / (den + 1e-8f);
      dturb   += stop + 0.05f * se + 0.01f * le;
      senergy += se;
    }
    out2[0] = dturb * (1.0f / 6.0f);
    out2[1] = senergy * (1.0f / 6.0f);
  }
}

// ---------------- host launcher ----------------
extern "C" void kernel_launch(void* const* d_in, const int* in_sizes, int n_in,
                              void* d_out, int out_size, void* d_ws, size_t ws_size,
                              hipStream_t stream) {
  const float* u0  = (const float*)d_in[0];
  const float* w1  = (const float*)d_in[1];
  const float* b1  = (const float*)d_in[2];
  const float* w2  = (const float*)d_in[3];
  const float* b2  = (const float*)d_in[4];
  const float* gw  = (const float*)d_in[5];
  const float* gb  = (const float*)d_in[6];
  const float* vw  = (const float*)d_in[7];
  const float* vb  = (const float*)d_in[8];
  const float* nw  = (const float*)d_in[9];
  const float* lw  = (const float*)d_in[10];
  const float* lb  = (const float*)d_in[11];
  const float* ag  = (const float*)d_in[12];
  const float* al  = (const float*)d_in[13];
  const float* ldt = (const float*)d_in[14];
  const float* dwv = (const float*)d_in[15];

  float* out = (float*)d_out;
  float* ws  = (float*)d_ws;
  // workspace layout (floats, then halves)
  float* bufA   = ws;                              // NTOT_   (u ping buffer)
  float* react  = ws + (size_t)NTOT_;              // NTOT_
  float* pooled = ws + 2 * (size_t)NTOT_;          // 256
  float* local4 = pooled + 256;                    // 4096
  float* lconv  = local4 + 4096;                   // 4096
  float* hstate = lconv + 4096;                    // 256
  float* acc    = hstate + 256;                    // 32 (24 used)
  float* prevp  = acc + 32;                        // 16384
  float* curp   = prevp + NPROBE_;                 // 16384
  _Float16* u16 = (_Float16*)(curp + NPROBE_);     // NTOT_ halves (A-operand mirror)
  _Float16* w1t = u16 + (size_t)NTOT_;             // 8192 halves
  _Float16* w2t = w1t + HID_ * C_;                 // 8192 halves

  // one-time (per call) prep: reset state, pack weights, build f16 mirror, probe0
  zero_kernel<<<2, 256, 0, stream>>>(hstate, 256 + 32);
  pack_weights_kernel<<<(HID_ * C_ + 255) / 256, 256, 0, stream>>>(w1, w2, w1t, w2t);
  u16_init_kernel<<<NROWS_ / 256, 256, 0, stream>>>(u0, u16);
  probe_init_kernel<<<B_ * C_, 256, 0, stream>>>(u0, prevp);

  for (int s = 0; s < 6; ++s) {
    // ping-pong: s0: in -> bufA; odd s: bufA -> d_out; even s>0: d_out -> bufA
    const float* uin = (s == 0) ? u0 : ((s & 1) ? bufA : out);
    float* uout = (s & 1) ? out : bufA;

    zero_kernel<<<(NPROBE_ + 255) / 256, 256, 0, stream>>>(curp, NPROBE_);
    pool_kernel<<<B_ * C_, 256, 0, stream>>>(uin, pooled, local4);
    state_kernel<<<1, 256, 0, stream>>>(pooled, gw, gb, vw, vb, lw, lb, local4, hstate, lconv);
    react_kernel<<<NROWS_ / 128, 256, 0, stream>>>(u16, w1t, b1, w2t, b2, react);
    combine_kernel<<<NROWS_ / 256, 256, 0, stream>>>(uin, react, hstate, lconv, nw,
                                                     ag, al, ldt, dwv,
                                                     uout, u16, curp, acc, s,
                                                     ((s + 1) % 2 == 0) ? 1 : 0);
    probe_finish_kernel<<<(NPROBE_ + 255) / 256, 256, 0, stream>>>(curp, prevp, acc, s);
  }
  // final u is in d_out[0..NTOT_); append the two scalar outputs
  finalize_kernel<<<1, 32, 0, stream>>>(acc, out + NTOT_);
}
